// DAGLayer_37280316129534
// MI455X (gfx1250) — matrix operand order repaired and checked by reading, not compiled, not run on hardware
//
#include <hip/hip_runtime.h>
#include <hip/hip_bf16.h>

// ---------------- problem constants ----------------
#define MAXA        50
#define NGF         30
#define NAF         75
#define NATOMS      6400
#define HIDDEN      100

// GEMM1 re-layout: K = 96 (padded atom feats) + 49 * 32 (padded gf rows) = 1664
#define K1          1664
#define FEAT_KSTEPS 3          // 96 / 32
#define GF_KSTEPS   49
#define N1PAD       128        // 8 N-tiles of 16 (cols 100..127 zero)
// GEMM2: K padded 100->128 (4 k-steps), N padded 30->32 (2 tiles)
#define K2          128
#define N2PAD       32

#define ATOMS_PER_BLOCK 64
#define THREADS         256    // 8 wave32
#define NBLOCKS         (NATOMS / ATOMS_PER_BLOCK)   // 100

// ---- LDS layout (halves unless noted) ----
#define GF_ROW_H    32                      // one gf row: 30 valid + 2 zero pad = 64 B
#define GF_ATOM_H   1640                    // 51*32 = 1632, +8 pad -> 3280 B / atom
#define FEAT_H      104                     // 75 valid + pad -> 208 B / atom
#define H_STRIDE    136                     // 128 valid + pad -> 272 B / atom

#define GF_BYTES    (ATOMS_PER_BLOCK * GF_ATOM_H * 2)   // 209920
#define FEAT_BYTES  (ATOMS_PER_BLOCK * FEAT_H * 2)      // 13312
#define H_BYTES     (ATOMS_PER_BLOCK * H_STRIDE * 2)    // 17408
#define PAR_BYTES   (ATOMS_PER_BLOCK * MAXA * 4)        // 12800
#define SMEM_BYTES  (GF_BYTES + FEAT_BYTES + H_BYTES + PAR_BYTES + 2 * ATOMS_PER_BLOCK * 4)

#define W0T_BYTES   (N1PAD * K1 * 2)        // 425984
#define W1T_BYTES   (N2PAD * K2 * 2)        // 8192

typedef _Float16 v16h __attribute__((ext_vector_type(16)));
typedef _Float16 v8h  __attribute__((ext_vector_type(8)));
typedef float    v8f  __attribute__((ext_vector_type(8)));

#define WMMA_F16(A, B, C) __builtin_amdgcn_wmma_f32_16x16x32_f16( \
    false, (A), false, (B), (short)0, (C), false, false)

// A fragment (16-bit A 16x32): two 16B chunks, 32 B apart (K +0..7 / +16..23 per half-wave)
__device__ __forceinline__ v16h ldA(const _Float16* p) {
    union { v16h v; v8h h[2]; } u;
    u.h[0] = *(const v8h*)(p);
    u.h[1] = *(const v8h*)(p + 16);
    return u.v;
}
// B fragment: two 16B chunks, 32 B apart (K +0..7 / +16..23 per half-wave), this lane's column
__device__ __forceinline__ v16h ldB(const _Float16* p) {
    union { v16h v; v8h h[2]; } u;
    u.h[0] = *(const v8h*)(p);
    u.h[1] = *(const v8h*)(p + 16);
    return u.v;
}

// ---------------- prologue: weight transpose + f16 convert + K/N padding ----------------
__global__ void build_w0t(const float* __restrict__ W0, _Float16* __restrict__ W0T) {
    int idx = blockIdx.x * 256 + threadIdx.x;
    if (idx >= N1PAD * K1) return;
    int n = idx / K1, k = idx - n * K1;
    float v = 0.0f;
    if (n < HIDDEN) {
        if (k < 96) {
            if (k < NAF) v = W0[k * HIDDEN + n];
        } else {
            int j = (k - 96) >> 5, c = (k - 96) & 31;
            if (c < NGF) v = W0[(NAF + j * NGF + c) * HIDDEN + n];
        }
    }
    W0T[idx] = (_Float16)v;
}

__global__ void build_w1t(const float* __restrict__ W1, _Float16* __restrict__ W1T) {
    int idx = blockIdx.x * 256 + threadIdx.x;
    if (idx >= N2PAD * K2) return;
    int n = idx >> 7, k = idx & 127;
    float v = (n < NGF && k < HIDDEN) ? W1[k * NGF + n] : 0.0f;
    W1T[idx] = (_Float16)v;
}

// ---------------- main DAG recurrence kernel ----------------
// block: 64 atoms, 8 wave32. Wave pair per 16-atom M-tile; even wave: N-tiles 0..3,
// odd wave: N-tiles 4..7 (tile 7 is the zero pad tile, keeps control flow wave-uniform).
__global__ void __launch_bounds__(THREADS, 1)
dag_layer_kernel(const float* __restrict__ atom_features,
                 const int*   __restrict__ parents,
                 const int*   __restrict__ orders,
                 const int*   __restrict__ masks,
                 const _Float16* __restrict__ W0T,
                 const _Float16* __restrict__ W1T,
                 const float* __restrict__ b0,
                 const float* __restrict__ b1,
                 float* __restrict__ out)
{
    extern __shared__ __align__(16) char smem[];
    _Float16* gfl   = (_Float16*)(smem);
    _Float16* featl = (_Float16*)(smem + GF_BYTES);
    _Float16* hbuf  = (_Float16*)(smem + GF_BYTES + FEAT_BYTES);
    int*      parl  = (int*)(smem + GF_BYTES + FEAT_BYTES + H_BYTES);
    int*      ordl  = parl + ATOMS_PER_BLOCK * MAXA;
    int*      mskl  = ordl + ATOMS_PER_BLOCK;

    const int tid   = threadIdx.x;
    const int w     = tid >> 5;          // wave 0..7
    const int l     = tid & 31;          // lane
    const int m     = l & 15;            // row / column within 16-tile
    const int hi8   = (l >= 16) ? 8 : 0; // A-chunk half offset for upper half-wave
    const int hiB   = (l >= 16) ? 8 : 0; // B-chunk half offset
    const int mt    = w >> 1;            // M-tile 0..3
    const int tbase = (w & 1) ? 4 : 0;   // first N-tile of this wave
    const int gbase = blockIdx.x * ATOMS_PER_BLOCK;
    const int a     = mt * 16 + m;       // this lane's A-row atom (block-local)

    // ---- zero init: gf tables (reference starts from zeros) + feat pad cols ----
    {
        int* gz = (int*)gfl;
        for (int e = tid; e < ATOMS_PER_BLOCK * GF_ATOM_H / 2; e += THREADS) gz[e] = 0;
        for (int e = tid; e < ATOMS_PER_BLOCK * (FEAT_H - NAF); e += THREADS) {
            int aa = e / (FEAT_H - NAF), f = NAF + (e - aa * (FEAT_H - NAF));
            featl[aa * FEAT_H + f] = (_Float16)0.0f;
        }
    }

    // per-lane biases (invariant over steps)
    float b0r[4];
#pragma unroll
    for (int u = 0; u < 4; ++u) {
        int n = (tbase + u) * 16 + m;
        b0r[u] = (n < HIDDEN) ? b0[n] : 0.0f;
    }
    const int n2 = (w & 1) * 16 + m;     // GEMM2 output column
    const float b1r = (n2 < NGF) ? b1[n2] : 0.0f;

    // per-lane B row offsets for GEMM1 (halves)
    int rowoff[4];
#pragma unroll
    for (int u = 0; u < 4; ++u) rowoff[u] = ((tbase + u) * 16 + m) * K1;

    // ---- t-invariant fragments we deliberately keep resident in VGPRs ----
    // 12 feature-K B fragments (96 VGPRs) + 4 GEMM2 B fragments (32 VGPRs).
    v16h Bf[FEAT_KSTEPS][4];
#pragma unroll
    for (int kt = 0; kt < FEAT_KSTEPS; ++kt)
#pragma unroll
        for (int u = 0; u < 4; ++u)
            Bf[kt][u] = ldB(W0T + rowoff[u] + kt * 32 + hiB);

    v16h B2f[4];
#pragma unroll
    for (int kt = 0; kt < 4; ++kt)
        B2f[kt] = ldB(W1T + (size_t)n2 * K2 + kt * 32 + hiB);

    __syncthreads();

#pragma unroll 1
    for (int t = 0; t < MAXA; ++t) {
        // Launder a 32-bit ZERO offset (not the pointer!) so the gf-K B loads below
        // are not loop-invariant -> no cross-t hoisting/spilling, while the address
        // stays derived from W0T (known global address space -> global_load_b128,
        // LOADcnt only, no false DScnt coupling with the LDS A-fragment loads).
        int lz = 0;
        asm volatile("" : "+v"(lz));
        const _Float16* w0t_j = W0T + lz + 96 + hiB;

        // ---- stage 1: load this step's order / mask / parent rows ----
        for (int e = tid; e < ATOMS_PER_BLOCK; e += THREADS) {
            ordl[e] = orders[(gbase + e) * MAXA + t];
            mskl[e] = masks[(gbase + e) * MAXA + t];
        }
        for (int e = tid; e < ATOMS_PER_BLOCK * MAXA; e += THREADS) {
            int aa = e / MAXA, j = e - aa * MAXA;
            parl[e] = parents[(size_t)(gbase + aa) * (MAXA * MAXA) + t * MAXA + j];
        }
        // prefetch next step's parent rows (1 cacheline / thread covers the slab)
        if (t + 1 < MAXA) {
            int aa = tid >> 1, half = tid & 1;
            __builtin_prefetch(
                (const char*)(parents + (size_t)(gbase + aa) * (MAXA * MAXA) + (t + 1) * MAXA)
                    + half * 128, 0, 0);
        }
        __syncthreads();

        // ---- stage 2: gather atom features (f32 -> f16) ----
        for (int e = tid; e < ATOMS_PER_BLOCK * NAF; e += THREADS) {
            int aa = e / NAF, f = e - aa * NAF;
            featl[aa * FEAT_H + f] = (_Float16)atom_features[(size_t)ordl[aa] * NAF + f];
        }
        __syncthreads();

        // ---- stage 3: GEMM1  h = relu(x @ W0 + b0), x gathered on the fly ----
        v8f acc[4];
#pragma unroll
        for (int u = 0; u < 4; ++u) acc[u] = (v8f){0,0,0,0,0,0,0,0};

        // feature K-steps (K = 0..95); B fragments live in registers
#pragma unroll
        for (int kt = 0; kt < FEAT_KSTEPS; ++kt) {
            v16h A = ldA(featl + a * FEAT_H + kt * 32 + hi8);
            acc[0] = WMMA_F16(A, Bf[kt][0], acc[0]);
            acc[1] = WMMA_F16(A, Bf[kt][1], acc[1]);
            acc[2] = WMMA_F16(A, Bf[kt][2], acc[2]);
            acc[3] = WMMA_F16(A, Bf[kt][3], acc[3]);
        }

        // gathered gf K-steps: one parent row per K-step (K = 96..1663).
        // Batch the 4 B-fragment loads ahead of the WMMAs and software-pipeline the
        // parent-index + A-fragment LDS load so global latency overlaps the WMMA chain.
        const int* parow = parl + a * MAXA + 1;
        const _Float16* gfatom = gfl + a * GF_ATOM_H + hi8;
        v16h Acur = ldA(gfatom + parow[0] * GF_ROW_H);
#pragma unroll 2
        for (int j = 0; j < GF_KSTEPS; ++j) {
            const _Float16* bp = w0t_j + j * 32;
            v16h B0 = ldB(bp + rowoff[0]);
            v16h B1 = ldB(bp + rowoff[1]);
            v16h B2 = ldB(bp + rowoff[2]);
            v16h B3 = ldB(bp + rowoff[3]);
            v16h Anxt = Acur;
            if (j + 1 < GF_KSTEPS) Anxt = ldA(gfatom + parow[j + 1] * GF_ROW_H);
            acc[0] = WMMA_F16(Acur, B0, acc[0]);
            acc[1] = WMMA_F16(Acur, B1, acc[1]);
            acc[2] = WMMA_F16(Acur, B2, acc[2]);
            acc[3] = WMMA_F16(Acur, B3, acc[3]);
            Acur = Anxt;
        }

        // bias + relu, store h tile as f16 (C layout: lane=N col, VGPR r = M row (+8 hi))
#pragma unroll
        for (int u = 0; u < 4; ++u) {
            const int n = (tbase + u) * 16 + m;
#pragma unroll
            for (int r = 0; r < 8; ++r) {
                float hv = acc[u][r] + b0r[u];
                hv = hv > 0.0f ? hv : 0.0f;
                int am = mt * 16 + r + hi8;
                hbuf[am * H_STRIDE + n] = (_Float16)hv;
            }
        }
        __syncthreads();

        // ---- stage 4: GEMM2  o = relu(h @ W1 + b1); B fragments live in registers ----
        v16h A2[4];
#pragma unroll
        for (int kt = 0; kt < 4; ++kt)
            A2[kt] = ldA(hbuf + (mt * 16 + m) * H_STRIDE + kt * 32 + hi8);
        v8f acc2 = (v8f){0,0,0,0,0,0,0,0};
#pragma unroll
        for (int kt = 0; kt < 4; ++kt)
            acc2 = WMMA_F16(A2[kt], B2f[kt], acc2);

        // ---- stage 5: masked gf-row update / final masked output ----
        if (t < MAXA - 1) {
#pragma unroll
            for (int r = 0; r < 8; ++r) {
                float v = acc2[r] + b1r;
                v = v > 0.0f ? v : 0.0f;
                int am = mt * 16 + r + hi8;
                if (n2 < NGF && mskl[am]) {
                    int col = parl[am * MAXA];            // par[i, t, 0]
                    gfl[am * GF_ATOM_H + col * GF_ROW_H + n2] = (_Float16)v;
                }
            }
        } else {
#pragma unroll
            for (int r = 0; r < 8; ++r) {
                float v = acc2[r] + b1r;
                v = v > 0.0f ? v : 0.0f;
                int am = mt * 16 + r + hi8;
                if (n2 < NGF)
                    out[(size_t)(gbase + am) * NGF + n2] = mskl[am] ? v : 0.0f;
            }
        }
        __syncthreads();
    }
}

// ---------------- host entry ----------------
extern "C" void kernel_launch(void* const* d_in, const int* in_sizes, int n_in,
                              void* d_out, int out_size, void* d_ws, size_t ws_size,
                              hipStream_t stream) {
    const float* atom_features = (const float*)d_in[0];
    const int*   parents       = (const int*)d_in[1];
    const int*   orders        = (const int*)d_in[2];
    const int*   masks         = (const int*)d_in[3];   // bool mask passed as int per harness
    const float* W0            = (const float*)d_in[5];
    const float* b0            = (const float*)d_in[6];
    const float* W1            = (const float*)d_in[7];
    const float* b1            = (const float*)d_in[8];
    float*       out           = (float*)d_out;

    _Float16* W0T = (_Float16*)d_ws;
    _Float16* W1T = (_Float16*)((char*)d_ws + W0T_BYTES);

    build_w0t<<<(N1PAD * K1 + 255) / 256, 256, 0, stream>>>(W0, W0T);
    build_w1t<<<(32 * 128 + 255) / 256, 256, 0, stream>>>(W1, W1T);

    (void)hipFuncSetAttribute((const void*)dag_layer_kernel,
                              hipFuncAttributeMaxDynamicSharedMemorySize,
                              (int)SMEM_BYTES);
    dag_layer_kernel<<<NBLOCKS, THREADS, SMEM_BYTES, stream>>>(
        atom_features, parents, orders, masks, W0T, W1T, b0, b1, out);
}
